// Net_82738249990826
// MI455X (gfx1250) — matrix operand order repaired and checked by reading, not compiled, not run on hardware
//
#include <hip/hip_runtime.h>
#include <hip/hip_bf16.h>

// ---------------- problem constants ----------------
constexpr int G    = 64;
constexpr int NPG  = 2048;
constexpr int DD   = 128;          // feature dim
constexpr int NN   = G * NPG;      // 131072 nodes
constexpr int DEG  = 16;
constexpr int EE   = NN * DEG;     // 2097152 edges
constexpr int K1   = 1639;         // ceil(0.8*2048)
constexpr int K2   = 1312;         // ceil(0.8*1639)
constexpr int K3   = 1050;         // ceil(0.8*1312)
constexpr int SORTN = 2048;

typedef __attribute__((ext_vector_type(16))) _Float16 v16h;
typedef __attribute__((ext_vector_type(8)))  _Float16 v8h;
typedef __attribute__((ext_vector_type(8)))  float    v8f;
typedef __attribute__((ext_vector_type(4)))  unsigned u32x4;
typedef __attribute__((ext_vector_type(4)))  int      i32x4;
typedef __attribute__((ext_vector_type(8)))  int      i32x8;

// ---------------- small utility kernels ----------------
__global__ void fill_f32(float* p, float v, int n) {
    int i = blockIdx.x * blockDim.x + threadIdx.x;
    if (i < n) p[i] = v;
}
__global__ void fill_i32(int* p, int v, int n) {
    int i = blockIdx.x * blockDim.x + threadIdx.x;
    if (i < n) p[i] = v;
}

// x[i][:] = emb[node_ids[i]][:]   (32 lanes per node, 4 floats each)
__global__ void embed_k(const int* __restrict__ ids, const float* __restrict__ emb,
                        float* __restrict__ x, int n) {
    int t = blockIdx.x * blockDim.x + threadIdx.x;
    int i = t >> 5, c = (t & 31) * 4;
    if (i >= n) return;
    int r = ids[i];
    float4 v = *(const float4*)(emb + (size_t)r * DD + c);
    *(float4*)(x + (size_t)i * DD + c) = v;
}

__global__ void edge_init_k(const int* __restrict__ ei, int* src, int* dst, int* em, int e) {
    int i = blockIdx.x * blockDim.x + threadIdx.x;
    if (i >= e) return;
    src[i] = ei[i];
    dst[i] = ei[EE + i];
    em[i]  = 1;
}

// scatter-add features of src into agg[dst]; count in-degree. 32 lanes / edge.
__global__ void aggregate_k(const int* __restrict__ src, const int* __restrict__ dst,
                            const int* __restrict__ em, const float* __restrict__ x,
                            float* __restrict__ agg, float* __restrict__ cnt, int e) {
    int t = blockIdx.x * blockDim.x + threadIdx.x;
    int i = t >> 5, c = (t & 31) * 4;
    if (i >= e) return;
    if (!em[i]) return;
    int s = src[i], d = dst[i];
    float4 v = *(const float4*)(x + (size_t)s * DD + c);
    float* a = agg + (size_t)d * DD + c;
    atomicAdd(a + 0, v.x);
    atomicAdd(a + 1, v.y);
    atomicAdd(a + 2, v.z);
    atomicAdd(a + 3, v.w);
    if ((t & 31) == 0) atomicAdd(cnt + d, 1.0f);
}

// pack weights: wcat[k][n] = (k<128 ? Wl[n][k] : Wr[n][k-128]), f16, row-major K x N
__global__ void wcat_build_k(const float* __restrict__ Wl, const float* __restrict__ Wr,
                             _Float16* __restrict__ wcat) {
    int t = blockIdx.x * blockDim.x + threadIdx.x;
    if (t >= 256 * 128) return;
    int k = t >> 7, n = t & 127;
    float v = (k < 128) ? Wl[n * 128 + k] : Wr[n * 128 + (k - 128)];
    wcat[t] = (_Float16)v;
}

// ---------------- TDM: stage the 256x128 f16 weight panel into LDS ----------------
// D# per CDNA5 ISA: group0 {count=1, lds_addr, global_addr, type=2},
// group1 {data_size=1(2B), tensor_dim0=128, tensor_dim1=256, tile 128x256, stride0=128}.
__device__ __forceinline__ void tdm_load_b_panel(const _Float16* gsrc, unsigned lds_off) {
#if __has_builtin(__builtin_amdgcn_tensor_load_to_lds)
    size_t ga = (size_t)gsrc;
    u32x4 g0;
    g0[0] = 1u;                                            // count=1, user desc
    g0[1] = lds_off;                                       // lds_addr (bytes)
    g0[2] = (unsigned)(ga & 0xFFFFFFFFu);                  // global_addr[31:0]
    g0[3] = (unsigned)((ga >> 32) & 0x01FFFFFFu) | (2u << 30); // addr[56:32] | type=2
    i32x8 g1;
    g1[0] = (1 << 16);       // workgroup_mask=0, data_size=1 (2 bytes)
    g1[1] = (128 << 16);     // tensor_dim0[15:0] = 128  (bits 79:48 start at dw1[31:16])
    g1[2] = (256 << 16);     // tensor_dim0 hi = 0, tensor_dim1[15:0] = 256
    g1[3] = (128 << 16);     // tensor_dim1 hi = 0, tile_dim0 = 128
    g1[4] = 256;             // tile_dim1 = 256, tile_dim2 = 0
    g1[5] = 128;             // tensor_dim0_stride lo32 = 128 elements
    g1[6] = 0;               // stride0 hi / stride1 lo
    g1[7] = 0;               // stride1 hi
    i32x4 z4 = {0, 0, 0, 0};
#if defined(__clang_major__) && (__clang_major__ >= 23)
    i32x8 z8 = {0, 0, 0, 0, 0, 0, 0, 0};
    __builtin_amdgcn_tensor_load_to_lds(g0, g1, z4, z4, z8, 0);
#else
    __builtin_amdgcn_tensor_load_to_lds(g0, g1, z4, z4, 0);
#endif
#endif
}

// ---------------- WMMA SAGE GEMM ----------------
// out = relu([mean | x] @ wcat + bias), mean = agg/max(cnt,1)
// block: 128 threads = 4 waves; each wave owns a 16-row M tile; block covers 64 rows.
// nrows is always a multiple of 64 (G=64 graphs * k nodes), grid is exact: no guards.
__global__ __launch_bounds__(128) void sage_gemm_k(
    const float* __restrict__ x, const float* __restrict__ agg,
    const float* __restrict__ cnt, const _Float16* __restrict__ wcat,
    const float* __restrict__ bias, float* __restrict__ out) {

    __shared__ _Float16 bsh[256 * 128];    // 64 KB weight panel

    const int wave = threadIdx.x >> 5;
    const int lane = threadIdx.x & 31;
    const int g    = lane >> 4;            // half-wave select
    const int l16  = lane & 15;
    const int m0   = blockIdx.x * 64 + wave * 16;
    const int mrow = m0 + l16;

#if __has_builtin(__builtin_amdgcn_tensor_load_to_lds)
    if (threadIdx.x == 0) {
        tdm_load_b_panel(wcat, (unsigned)(size_t)&bsh[0]);
    }
#if __has_builtin(__builtin_amdgcn_s_wait_tensorcnt)
    __builtin_amdgcn_s_wait_tensorcnt(0);
#endif
    __syncthreads();
#else
    // cooperative fallback copy global -> LDS
    for (int i = threadIdx.x; i < 256 * 128 / 8; i += 128)
        ((v8h*)bsh)[i] = ((const v8h*)wcat)[i];
    __syncthreads();
#endif

    const float cdeg = cnt[mrow];
    const float inv  = 1.0f / fmaxf(cdeg, 1.0f);
    const float* arow = agg + (size_t)mrow * DD;
    const float* xrow = x   + (size_t)mrow * DD;

    // A fragments: 16x32 f16 layout. lane l: row M = l&15.
    // halves 0..7  <- K = kb*32 + g*8 + j ; halves 8..15 <- K = kb*32 + 16 + g*8 + j
    v16h afrag[8];
#pragma unroll
    for (int kb = 0; kb < 8; ++kb) {
#pragma unroll
        for (int h = 0; h < 2; ++h) {
            int kofs = kb * 32 + h * 16 + g * 8;
            const float* sp; float sc;
            if (kofs < 128) { sp = arow + kofs;         sc = inv;  }
            else            { sp = xrow + (kofs - 128); sc = 1.0f; }
            float4 v0 = *(const float4*)(sp);
            float4 v1 = *(const float4*)(sp + 4);
            int b = h * 8;
            afrag[kb][b + 0] = (_Float16)(v0.x * sc);
            afrag[kb][b + 1] = (_Float16)(v0.y * sc);
            afrag[kb][b + 2] = (_Float16)(v0.z * sc);
            afrag[kb][b + 3] = (_Float16)(v0.w * sc);
            afrag[kb][b + 4] = (_Float16)(v1.x * sc);
            afrag[kb][b + 5] = (_Float16)(v1.y * sc);
            afrag[kb][b + 6] = (_Float16)(v1.z * sc);
            afrag[kb][b + 7] = (_Float16)(v1.w * sc);
        }
    }

    // loop the 8 N tiles (output cols), K-loop of 8 x 32; B fragments come from LDS
#pragma unroll 1
    for (int nt = 0; nt < 8; ++nt) {
        v8f acc = {};
#pragma unroll
        for (int kb = 0; kb < 8; ++kb) {
            // B fragment: lane holds N-col = nt*16 + l16; halves 0..7 <- K = kb*32 + g*8 + q, halves 8..15 <- K = kb*32 + 16 + g*8 + q
            int krow = kb * 32 + g * 8;
            const _Float16* bp = bsh + (size_t)krow * 128 + nt * 16 + l16;
            v16h bfrag;
#pragma unroll
            for (int q = 0; q < 8; ++q) { bfrag[q] = bp[q * 128]; bfrag[8 + q] = bp[(16 + q) * 128]; }
            acc = __builtin_amdgcn_wmma_f32_16x16x32_f16(
                false, afrag[kb], false, bfrag, (short)0, acc, false, false);
        }
        // D layout: VGPR r, lane l -> row m0 + r + g*8, col nt*16 + l16
        float bn = bias[nt * 16 + l16];
#pragma unroll
        for (int r = 0; r < 8; ++r) {
            int row = m0 + r + g * 8;
            float v = acc[r] + bn;
            out[(size_t)row * DD + nt * 16 + l16] = fmaxf(v, 0.0f);
        }
    }
}

// ---------------- top-k pooling ----------------
__global__ void pwnorm_k(const float* __restrict__ pw, float* __restrict__ nrm) {
    __shared__ float s[128];
    int t = threadIdx.x;
    float v = pw[t];
    s[t] = v * v;
    __syncthreads();
    for (int o = 64; o > 0; o >>= 1) { if (t < o) s[t] += s[t + o]; __syncthreads(); }
    if (t == 0) *nrm = sqrtf(s[0]);
}

// score[i] = tanh(dot(x_i, pw)/||pw||); one wave per node
__global__ void score_k(const float* __restrict__ x, const float* __restrict__ pw,
                        const float* __restrict__ nrm, float* __restrict__ score, int n) {
    int t = blockIdx.x * blockDim.x + threadIdx.x;
    int i = t >> 5, lane = t & 31;
    if (i >= n) return;
    float4 a = *(const float4*)(x + (size_t)i * DD + lane * 4);
    float4 w = *(const float4*)(pw + lane * 4);
    float d = a.x * w.x + a.y * w.y + a.z * w.z + a.w * w.w;
    for (int m = 16; m > 0; m >>= 1) d += __shfl_xor(d, m, 32);
    if (lane == 0) score[i] = tanhf(d / (*nrm));
}

// one block per graph: LDS bitonic sort (descending), emit kept set + remap table
__global__ __launch_bounds__(512) void topk_sort_k(
    const float* __restrict__ score, int* __restrict__ kept, float* __restrict__ ksc,
    int* __restrict__ new_of_old, int n_per, int k) {
    __shared__ float s[SORTN];
    __shared__ int   idx[SORTN];
    int g = blockIdx.x, t = threadIdx.x;
    for (int j = t; j < SORTN; j += 512) {
        if (j < n_per) { s[j] = score[g * n_per + j]; idx[j] = j; }
        else           { s[j] = -1e30f;               idx[j] = -1; }
    }
    __syncthreads();
    for (int size = 2; size <= SORTN; size <<= 1) {
        for (int stride = size >> 1; stride > 0; stride >>= 1) {
            for (int u = t; u < SORTN / 2; u += 512) {
                int i = (u / stride) * (stride * 2) + (u % stride);
                int j = i + stride;
                bool desc = ((i & size) == 0);
                float si = s[i], sj = s[j];
                bool sw = desc ? (si < sj) : (si > sj);
                if (sw) {
                    s[i] = sj; s[j] = si;
                    int ti = idx[i]; idx[i] = idx[j]; idx[j] = ti;
                }
            }
            __syncthreads();
        }
    }
    for (int j = t; j < k; j += 512) {
        int oldg = g * n_per + idx[j];
        int newg = g * k + j;
        kept[newg] = oldg;
        ksc[newg]  = s[j];
        new_of_old[oldg] = newg;
    }
}

// xout[new][:] = xin[kept[new]][:] * ksc[new]
__global__ void pool_apply_k(const float* __restrict__ xin, const int* __restrict__ kept,
                             const float* __restrict__ ksc, float* __restrict__ xout, int n_new) {
    int t = blockIdx.x * blockDim.x + threadIdx.x;
    int i = t >> 5, c = (t & 31) * 4;
    if (i >= n_new) return;
    int o = kept[i]; float sc = ksc[i];
    float4 v = *(const float4*)(xin + (size_t)o * DD + c);
    v.x *= sc; v.y *= sc; v.z *= sc; v.w *= sc;
    *(float4*)(xout + (size_t)i * DD + c) = v;
}

__global__ void remap_k(int* src, int* dst, int* em, const int* __restrict__ new_of_old,
                        int e, int n_new) {
    int i = blockIdx.x * blockDim.x + threadIdx.x;
    if (i >= e) return;
    if (!em[i]) return;
    int sn = new_of_old[src[i]], dn = new_of_old[dst[i]];
    bool ok = (sn < n_new) && (dn < n_new);
    em[i]  = ok ? 1 : 0;
    src[i] = ok ? sn : 0;
    dst[i] = ok ? dn : 0;
}

// hsum[g] += cat(max_j x[g,j,:], mean_j x[g,j,:])
__global__ void readout_k(const float* __restrict__ x, float* __restrict__ hsum, int k) {
    int g = blockIdx.x, d = threadIdx.x;   // 128 threads
    float mx = -1e30f, sm = 0.0f;
    const float* base = x + (size_t)g * k * DD + d;
    for (int j = 0; j < k; ++j) {
        float v = base[(size_t)j * DD];
        mx = fmaxf(mx, v);
        sm += v;
    }
    hsum[g * 256 + d]       += mx;
    hsum[g * 256 + 128 + d] += sm / (float)k;
}

// ---------------- final MLP ----------------
__global__ void mlp1_k(const float* __restrict__ h, const float* __restrict__ W,
                       const float* __restrict__ b, float* __restrict__ o1) {
    int t = blockIdx.x * blockDim.x + threadIdx.x;
    if (t >= 64 * 128) return;
    int g = t >> 7, o = t & 127;
    float a = b[o];
    for (int k = 0; k < 256; ++k) a += h[g * 256 + k] * W[o * 256 + k];
    o1[t] = fmaxf(a, 0.0f);
}
__global__ void mlp2_k(const float* __restrict__ h, const float* __restrict__ W,
                       const float* __restrict__ b, float* __restrict__ o2) {
    int t = blockIdx.x * blockDim.x + threadIdx.x;
    if (t >= 64 * 64) return;
    int g = t >> 6, o = t & 63;
    float a = b[o];
    for (int k = 0; k < 128; ++k) a += h[g * 128 + k] * W[o * 128 + k];
    o2[t] = fmaxf(a, 0.0f);
}
__global__ void mlp3_k(const float* __restrict__ h, const float* __restrict__ W,
                       const float* __restrict__ b, float* __restrict__ out) {
    int g = threadIdx.x;
    if (g >= 64) return;
    float a = b[0];
    for (int k = 0; k < 64; ++k) a += h[g * 64 + k] * W[k];
    out[g] = 1.0f / (1.0f + expf(-a));
}

// ---------------- host-side orchestration ----------------
static inline int cdiv(long long a, long long b) { return (int)((a + b - 1) / b); }

extern "C" void kernel_launch(void* const* d_in, const int* in_sizes, int n_in,
                              void* d_out, int out_size, void* d_ws, size_t ws_size,
                              hipStream_t stream) {
    const int*   node_ids = (const int*)  d_in[0];
    const int*   edge_idx = (const int*)  d_in[1];
    const float* emb      = (const float*)d_in[3];
    const float* Wl[3] = { (const float*)d_in[4],  (const float*)d_in[8],  (const float*)d_in[12] };
    const float* Wr[3] = { (const float*)d_in[5],  (const float*)d_in[9],  (const float*)d_in[13] };
    const float* bl[3] = { (const float*)d_in[6],  (const float*)d_in[10], (const float*)d_in[14] };
    const float* pw[3] = { (const float*)d_in[7],  (const float*)d_in[11], (const float*)d_in[15] };
    const float* W1 = (const float*)d_in[16]; const float* b1 = (const float*)d_in[17];
    const float* W2 = (const float*)d_in[18]; const float* b2 = (const float*)d_in[19];
    const float* W3 = (const float*)d_in[20]; const float* b3 = (const float*)d_in[21];
    float* out = (float*)d_out;

    // workspace carve-up (256B aligned)
    char* ws = (char*)d_ws;
    size_t off = 0;
    auto alloc = [&](size_t bytes) { size_t o = off; off = (off + bytes + 255) & ~(size_t)255; return ws + o; };
    float*    bufX   = (float*)   alloc((size_t)NN * DD * 4);
    float*    bufY   = (float*)   alloc((size_t)NN * DD * 4);
    float*    agg    = (float*)   alloc((size_t)NN * DD * 4);
    float*    cnt    = (float*)   alloc((size_t)NN * 4);
    float*    score  = (float*)   alloc((size_t)NN * 4);
    int*      kept   = (int*)     alloc((size_t)NN * 4);
    float*    ksc    = (float*)   alloc((size_t)NN * 4);
    int*      noo    = (int*)     alloc((size_t)NN * 4);
    int*      srcb   = (int*)     alloc((size_t)EE * 4);
    int*      dstb   = (int*)     alloc((size_t)EE * 4);
    int*      emk    = (int*)     alloc((size_t)EE * 4);
    _Float16* wcat   = (_Float16*)alloc((size_t)3 * 256 * 128 * 2);
    float*    norms  = (float*)   alloc(3 * 4);
    float*    hsum   = (float*)   alloc(64 * 256 * 4);
    float*    h1     = (float*)   alloc(64 * 128 * 4);
    float*    h2     = (float*)   alloc(64 * 64 * 4);

    const int T = 256;

    // init
    fill_f32<<<cdiv(64 * 256, T), T, 0, stream>>>(hsum, 0.0f, 64 * 256);
    embed_k<<<cdiv((long long)NN * 32, T), T, 0, stream>>>(node_ids, emb, bufX, NN);
    edge_init_k<<<cdiv(EE, T), T, 0, stream>>>(edge_idx, srcb, dstb, emk, EE);
    for (int l = 0; l < 3; ++l) {
        wcat_build_k<<<cdiv(256 * 128, T), T, 0, stream>>>(Wl[l], Wr[l], wcat + (size_t)l * 256 * 128);
        pwnorm_k<<<1, 128, 0, stream>>>(pw[l], norms + l);
    }

    const int n_per[3] = { NPG, K1, K2 };
    const int kk[3]    = { K1,  K2, K3 };

    for (int l = 0; l < 3; ++l) {
        const int n_cur = G * n_per[l];   // always a multiple of 64
        const int n_new = G * kk[l];

        // mean aggregation
        fill_f32<<<cdiv((long long)n_cur * DD, T), T, 0, stream>>>(agg, 0.0f, n_cur * DD);
        fill_f32<<<cdiv(n_cur, T), T, 0, stream>>>(cnt, 0.0f, n_cur);
        aggregate_k<<<cdiv((long long)EE * 32, T), T, 0, stream>>>(srcb, dstb, emk, bufX, agg, cnt, EE);

        // SAGE GEMM (WMMA, TDM-staged weights) + bias + relu -> bufY
        sage_gemm_k<<<n_cur / 64, 128, 0, stream>>>(
            bufX, agg, cnt, wcat + (size_t)l * 256 * 128, bl[l], bufY);

        // top-k pooling
        score_k<<<cdiv((long long)n_cur * 32, T), T, 0, stream>>>(bufY, pw[l], norms + l, score, n_cur);
        fill_i32<<<cdiv(n_cur, T), T, 0, stream>>>(noo, n_new, n_cur);
        topk_sort_k<<<G, 512, 0, stream>>>(score, kept, ksc, noo, n_per[l], kk[l]);
        pool_apply_k<<<cdiv((long long)n_new * 32, T), T, 0, stream>>>(bufY, kept, ksc, bufX, n_new);
        remap_k<<<cdiv(EE, T), T, 0, stream>>>(srcb, dstb, emk, noo, EE, n_new);

        // readout accumulation
        readout_k<<<G, 128, 0, stream>>>(bufX, hsum, kk[l]);
    }

    // final MLP + sigmoid
    mlp1_k<<<cdiv(64 * 128, T), T, 0, stream>>>(hsum, W1, b1, h1);
    mlp2_k<<<cdiv(64 * 64, T), T, 0, stream>>>(h1, W2, b2, h2);
    mlp3_k<<<1, 64, 0, stream>>>(h2, W3, b3, out);
}
